// GATGFN_55379308315251
// MI455X (gfx1250) — matrix operand order repaired and checked by reading, not compiled, not run on hardware
//
#include <hip/hip_runtime.h>
#include <hip/hip_bf16.h>

typedef __attribute__((ext_vector_type(16))) _Float16 v16h;
typedef __attribute__((ext_vector_type(8)))  float    v8f;

#define NNODES 20000
#define NEDGES 500000
#define NBATCH 2
#define DD     128
#define NH     4
#define NL     3
#define HD     (NH * DD)        // 512
#define LD     (NL * DD)        // 384
#define LH1    (NL * NH + 1)    // 13
#define NEGSL  0.2f

union U16x16 { v16h h; unsigned u[8]; };
union HPack  { unsigned u; _Float16 h[2]; };

// ---------- order-preserving float <-> uint for atomicMax on floats ----------
__device__ __forceinline__ unsigned f2ord(float f) {
    unsigned u = __float_as_uint(f);
    return (u & 0x80000000u) ? ~u : (u | 0x80000000u);
}
__device__ __forceinline__ float ord2f(unsigned k) {
    unsigned u = (k & 0x80000000u) ? (k ^ 0x80000000u) : ~k;
    return __uint_as_float(u);
}

// =====================================================================
// 1) input embedding: h[n][i] = sum_k x[n][k] * W_in[i][k] + b_in[i]
//    A tile (16 rows x 128 K, f16-packed) staged once in LDS per block;
//    8 waves cover 16 rows x 128 cols; B pairs loaded as float2.
// =====================================================================
__global__ void k_gemm_in(const float* __restrict__ x, const float* __restrict__ Win,
                          const float* __restrict__ bin, float* __restrict__ h)
{
    __shared__ unsigned Ah[16 * 64];           // [m][kpair] packed half2
    const int tid = threadIdx.x;
    const int row0 = blockIdx.x * 16;
    for (int i = tid; i < 16 * 64; i += 256) {
        const int m = i >> 6, kp = i & 63;
        const float2 v = *(const float2*)&x[(size_t)(row0 + m) * DD + 2 * kp];
        HPack p; p.h[0] = (_Float16)v.x; p.h[1] = (_Float16)v.y;
        Ah[i] = p.u;
    }
    __syncthreads();

    const int t  = tid & 31, w = tid >> 5;
    const int hi = t >> 4,  mn = t & 15;
    const int col0 = w * 16;
    v8f c = {};
    for (int kb = 0; kb < DD; kb += 32) {
        U16x16 a, b;
#pragma unroll
        for (int j = 0; j < 8; ++j) {
            const int ka = 2 * j + 16 * hi;                                // even
            a.u[j] = Ah[mn * 64 + ((kb + ka) >> 1)];
            const int kn = 2 * j + 16 * hi;                                // even
            const float2 bb = *(const float2*)&Win[(size_t)(col0 + mn) * DD + kb + kn];
            HPack p; p.h[0] = (_Float16)bb.x; p.h[1] = (_Float16)bb.y;
            b.u[j] = p.u;
        }
        c = __builtin_amdgcn_wmma_f32_16x16x32_f16(false, a.h, false, b.h, (short)0, c, false, false);
    }
#pragma unroll
    for (int v = 0; v < 8; ++v) {
        const int m = v + 8 * hi;
        h[(size_t)(row0 + m) * DD + col0 + mn] = c[v] + bin[col0 + mn];
    }
}

// =====================================================================
// 2a) per-layer projection: hp[n][j] = sum_k h[n][k] * Wl[k][j], j<512
//     A (16x128) and B (128x128) tiles staged in LDS; WMMA frags are
//     contiguous ds_load_b32 of pre-packed half pairs.
// =====================================================================
__global__ void k_gemm_hp(const float* __restrict__ hcur, const float* __restrict__ Wl,
                          float* __restrict__ hp)
{
    __shared__ unsigned Ah[16 * 64];           // 4 KB : [m][kpair]
    __shared__ unsigned Bh[128 * 64];          // 32 KB: [col][kpair]
    const int tid = threadIdx.x;
    const int row0 = blockIdx.x * 16;
    const int cb0  = blockIdx.y * 128;

    for (int i = tid; i < 16 * 64; i += 256) {
        const int m = i >> 6, kp = i & 63;
        const float2 v = *(const float2*)&hcur[(size_t)(row0 + m) * DD + 2 * kp];
        HPack p; p.h[0] = (_Float16)v.x; p.h[1] = (_Float16)v.y;
        Ah[i] = p.u;
    }
    _Float16* Bhh = (_Float16*)Bh;             // half view [col][k]
    for (int i = tid; i < DD * 32; i += 256) { // 4096 float4 groups: 128 k x 32 colgroups
        const int k = i >> 5, colg = (i & 31) << 2;
        const float4 v = *(const float4*)&Wl[(size_t)k * HD + cb0 + colg];
        Bhh[(colg + 0) * DD + k] = (_Float16)v.x;
        Bhh[(colg + 1) * DD + k] = (_Float16)v.y;
        Bhh[(colg + 2) * DD + k] = (_Float16)v.z;
        Bhh[(colg + 3) * DD + k] = (_Float16)v.w;
    }
    __syncthreads();

    const int t  = tid & 31, w = tid >> 5;
    const int hi = t >> 4,  mn = t & 15;
    const int cl = w * 16 + mn;                // col within the 128-wide tile
    v8f c = {};
    for (int kb = 0; kb < DD; kb += 32) {
        U16x16 a, b;
#pragma unroll
        for (int j = 0; j < 8; ++j) {
            const int ka = 2 * j + 16 * hi;
            a.u[j] = Ah[mn * 64 + ((kb + ka) >> 1)];
            const int kn = 2 * j + 16 * hi;
            b.u[j] = Bh[cl * 64 + ((kb + kn) >> 1)];
        }
        c = __builtin_amdgcn_wmma_f32_16x16x32_f16(false, a.h, false, b.h, (short)0, c, false, false);
    }
#pragma unroll
    for (int v = 0; v < 8; ++v)
        hp[(size_t)(row0 + v + 8 * hi) * HD + cb0 + cl] = c[v];
}

// =====================================================================
// 2b) attention coefficients: alpha_{src,dst}[n][h] = <hp[n,h,:], a_{s,d}[h,:]>
// =====================================================================
__global__ void k_alpha(const float* __restrict__ hp, const float* __restrict__ as,
                        const float* __restrict__ ad, float* __restrict__ asb,
                        float* __restrict__ adb)
{
    const int i = blockIdx.x * blockDim.x + threadIdx.x;
    if (i >= NNODES * NH) return;
    const int n = i >> 2, hh = i & 3;
    const float* row = hp + (size_t)n * HD + hh * DD;
    const float* va = as + hh * DD;
    const float* vd = ad + hh * DD;
    float s0 = 0.f, s1 = 0.f;
#pragma unroll 8
    for (int d = 0; d < DD; ++d) { const float v = row[d]; s0 += v * va[d]; s1 += v * vd[d]; }
    asb[i] = s0;
    adb[i] = s1;
}

// zero out_accum (N*HD) + seg_max/seg_sum (N*NH each). grid sized so stride hits once.
__global__ void k_init_layer(float* __restrict__ oa, unsigned* __restrict__ sm,
                             float* __restrict__ ss)
{
    const int i = blockIdx.x * blockDim.x + threadIdx.x;
    for (int idx = i; idx < NNODES * HD; idx += gridDim.x * blockDim.x) oa[idx] = 0.f;
    if (i < NNODES * NH) { sm[i] = 0u; ss[i] = 0.f; }
}

// 2c) edge scores + segment max (ordered-uint atomicMax)
__global__ void k_edge_max(const int* __restrict__ ei, const float* __restrict__ asb,
                           const float* __restrict__ adb, float* __restrict__ ebuf,
                           unsigned* __restrict__ sm)
{
    const int i = blockIdx.x * blockDim.x + threadIdx.x;
    if (i >= NEDGES * NH) return;
    const int e = i >> 2, hh = i & 3;
    const int s = ei[e], d = ei[NEDGES + e];
    float v = asb[s * NH + hh] + adb[d * NH + hh];
    v = (v > 0.f) ? v : NEGSL * v;
    ebuf[i] = v;
    atomicMax(&sm[d * NH + hh], f2ord(v));
}

// 2d) exp(e - max) + segment sum
__global__ void k_edge_expsum(const int* __restrict__ ei, float* __restrict__ ebuf,
                              const unsigned* __restrict__ sm, float* __restrict__ ss)
{
    const int i = blockIdx.x * blockDim.x + threadIdx.x;
    if (i >= NEDGES * NH) return;
    const int e = i >> 2, hh = i & 3;
    const int d = ei[NEDGES + e];
    const float m = ord2f(sm[d * NH + hh]);
    const float ex = __expf(ebuf[i] - m);
    ebuf[i] = ex;
    atomicAdd(&ss[d * NH + hh], ex);
}

// 2e) alpha = ex/denom; save into ac; scatter hp[src]*alpha into out_accum[dst]
__global__ void k_edge_aggr(const int* __restrict__ ei, const float* __restrict__ ebuf,
                            const float* __restrict__ ss, const float* __restrict__ hp,
                            float* __restrict__ oa, float* __restrict__ ac, int l)
{
    const int i = blockIdx.x * blockDim.x + threadIdx.x;
    if (i >= NEDGES * NH) return;
    const int e = i >> 2, hh = i & 3;
    const int s = ei[e], d = ei[NEDGES + e];
    const float denom = ss[d * NH + hh] + 1e-16f;
    const float alpha = ebuf[i] / denom;
    ac[(size_t)e * (NL * NH) + l * NH + hh] = alpha;
    const float* hsrc = hp + (size_t)s * HD + hh * DD;
    __builtin_prefetch(hsrc, 0, 0);   // global_prefetch_b8, L2-resident gather
    float* odst = oa + (size_t)d * HD + hh * DD;
    const float4* h4 = (const float4*)hsrc;
#pragma unroll 4
    for (int q = 0; q < DD / 4; ++q) {
        const float4 v = h4[q];
        atomicAdd(odst + 4 * q + 0, v.x * alpha);
        atomicAdd(odst + 4 * q + 1, v.y * alpha);
        atomicAdd(odst + 4 * q + 2, v.z * alpha);
        atomicAdd(odst + 4 * q + 3, v.w * alpha);
    }
}

// 2f) head-mean + bias -> xc slice; next h = leaky_relu
__global__ void k_finalize(const float* __restrict__ oa, const float* __restrict__ bl,
                           float* __restrict__ xc, float* __restrict__ h, int l)
{
    const int i = blockIdx.x * blockDim.x + threadIdx.x;
    if (i >= NNODES * DD) return;
    const int n = i >> 7, dd = i & 127;
    float s = 0.f;
#pragma unroll
    for (int hh = 0; hh < NH; ++hh) s += oa[(size_t)n * HD + hh * DD + dd];
    const float val = s * 0.25f + bl[dd];
    xc[(size_t)n * LD + l * DD + dd] = val;
    h[i] = (val > 0.f) ? val : NEGSL * val;
}

__global__ void k_zero16(float* __restrict__ p)
{
    if (threadIdx.x < 16) p[threadIdx.x] = 0.f;
}

// 3) x_logits mean over nodes (bias b_xa folded in at readout)
__global__ void k_xlogits(const float* __restrict__ xc, const float* __restrict__ Wxa,
                          float* __restrict__ xl)
{
    const int n = blockIdx.x * blockDim.x + threadIdx.x;
    if (n >= NNODES) return;
    float acc[LH1];
#pragma unroll
    for (int j = 0; j < LH1; ++j) acc[j] = 0.f;
    const float* xr = xc + (size_t)n * LD;
    for (int k = 0; k < LD; ++k) {
        const float xv = xr[k];
#pragma unroll
        for (int j = 0; j < LH1; ++j) acc[j] += Wxa[j * LD + k] * xv;
    }
    const float inv = 1.0f / (float)NNODES;
#pragma unroll
    for (int j = 0; j < LH1; ++j) atomicAdd(&xl[j], acc[j] * inv);
}

// =====================================================================
// 4) fused readout: rows=(b,E+1), v[13] -> relu(v@W_r1^T+b_r1) -> dot W_r2 + b_r2
//    16-row tile per block; 8 waves each own 16 of the 128 hidden units;
//    WMMA K padded 13->32; hid never hits memory (shuffle-reduced in regs)
// =====================================================================
__global__ void k_readout(const float* __restrict__ state, const float* __restrict__ ac,
                          const float* __restrict__ xl, const float* __restrict__ bxa,
                          const float* __restrict__ Wr1, const float* __restrict__ br1,
                          const float* __restrict__ Wr2, const float* __restrict__ br2,
                          float* __restrict__ out)
{
    __shared__ _Float16 As[16 * 32];
    __shared__ float rowsum[16];
    const int tid = threadIdx.x;
    const int batch = blockIdx.y;
    const int base = blockIdx.x * 16;        // row base within (E+1)
    for (int idx = tid; idx < 16 * 32; idx += 256) {
        const int m = idx >> 5, k = idx & 31;
        const int r = base + m;
        float v = 0.f;
        if (r <= NEDGES && k < LH1) {
            if (r == 0)       v = xl[k] + bxa[k];
            else if (k == 0)  v = state[(size_t)batch * NEDGES + (r - 1)];
            else              v = ac[(size_t)(r - 1) * (NL * NH) + (k - 1)];
        }
        As[m * 32 + k] = (_Float16)v;
    }
    if (tid < 16) rowsum[tid] = 0.f;
    __syncthreads();

    const int t = tid & 31, w = tid >> 5;
    const int hi = t >> 4, mn = t & 15;
    const int col = w * 16 + mn;             // hidden unit 0..127
    v16h a, b;
#pragma unroll
    for (int j = 0; j < 8; ++j) {
        const int ka = 2 * j + 16 * hi;
        a[2 * j]     = As[mn * 32 + ka];
        a[2 * j + 1] = As[mn * 32 + ka + 1];
        const int kn = 2 * j + 16 * hi;
        _Float16 b0 = (_Float16)0.f, b1 = (_Float16)0.f;
        if (kn < LH1)     b0 = (_Float16)Wr1[col * LH1 + kn];
        if (kn + 1 < LH1) b1 = (_Float16)Wr1[col * LH1 + kn + 1];
        b[2 * j] = b0; b[2 * j + 1] = b1;
    }
    v8f c = {};
    c = __builtin_amdgcn_wmma_f32_16x16x32_f16(false, a, false, b, (short)0, c, false, false);

    const float bias = br1[col];
    const float w2   = Wr2[col];
#pragma unroll
    for (int v = 0; v < 8; ++v) {
        float hv = c[v] + bias;
        hv = (hv > 0.f) ? hv : 0.f;
        float p = hv * w2;
        p += __shfl_xor(p, 8, 32);           // reduce across the 16 lanes of each half
        p += __shfl_xor(p, 4, 32);
        p += __shfl_xor(p, 2, 32);
        p += __shfl_xor(p, 1, 32);
        if (mn == 0) atomicAdd(&rowsum[v + 8 * hi], p);
    }
    __syncthreads();
    if (tid < 16) {
        const int r = base + tid;
        if (r <= NEDGES)
            out[(size_t)batch * (NEDGES + 1) + r] = rowsum[tid] + br2[0];
    }
}

// =====================================================================
extern "C" void kernel_launch(void* const* d_in, const int* in_sizes, int n_in,
                              void* d_out, int out_size, void* d_ws, size_t ws_size,
                              hipStream_t stream)
{
    const float* state = (const float*)d_in[0];
    const int*   ei    = (const int*)  d_in[1];
    const float* x     = (const float*)d_in[2];
    const float* Win   = (const float*)d_in[3];
    const float* bin   = (const float*)d_in[4];
    const float* Wl    = (const float*)d_in[5];   // (L, D, H*D)
    const float* asrc  = (const float*)d_in[6];   // (L, H, D)
    const float* adst  = (const float*)d_in[7];
    const float* bl    = (const float*)d_in[8];   // (L, D)
    const float* Wxa   = (const float*)d_in[9];   // (13, 384)
    const float* bxa   = (const float*)d_in[10];
    const float* Wr1   = (const float*)d_in[11];  // (128, 13)
    const float* br1   = (const float*)d_in[12];
    const float* Wr2   = (const float*)d_in[13];  // (1, 128)
    const float* br2   = (const float*)d_in[14];
    float* out = (float*)d_out;

    // workspace layout (floats)
    float*    ws    = (float*)d_ws;
    float*    h_buf = ws;                                      // N*D
    float*    hp    = h_buf + (size_t)NNODES * DD;             // N*HD
    float*    oa    = hp + (size_t)NNODES * HD;                // N*HD
    float*    asb   = oa + (size_t)NNODES * HD;                // N*NH
    float*    adb   = asb + (size_t)NNODES * NH;               // N*NH
    float*    ebuf  = adb + (size_t)NNODES * NH;               // E*NH
    unsigned* segm  = (unsigned*)(ebuf + (size_t)NEDGES * NH); // N*NH
    float*    segs  = (float*)(segm + (size_t)NNODES * NH);    // N*NH
    float*    xc    = segs + (size_t)NNODES * NH;              // N*L*D
    float*    acb   = xc + (size_t)NNODES * LD;                // E*L*NH
    float*    xlac  = acb + (size_t)NEDGES * NL * NH;          // 16

    k_gemm_in<<<NNODES / 16, 256, 0, stream>>>(x, Win, bin, h_buf);

    const int egrid = (NEDGES * NH + 255) / 256;
    for (int l = 0; l < NL; ++l) {
        k_gemm_hp<<<dim3(NNODES / 16, 4), 256, 0, stream>>>(h_buf, Wl + (size_t)l * DD * HD, hp);
        k_alpha<<<(NNODES * NH + 255) / 256, 256, 0, stream>>>(
            hp, asrc + (size_t)l * NH * DD, adst + (size_t)l * NH * DD, asb, adb);
        k_init_layer<<<(NNODES * HD) / 256, 256, 0, stream>>>(oa, segm, segs);
        k_edge_max<<<egrid, 256, 0, stream>>>(ei, asb, adb, ebuf, segm);
        k_edge_expsum<<<egrid, 256, 0, stream>>>(ei, ebuf, segm, segs);
        k_edge_aggr<<<egrid, 256, 0, stream>>>(ei, ebuf, segs, hp, oa, acb, l);
        k_finalize<<<(NNODES * DD + 255) / 256, 256, 0, stream>>>(oa, bl + (size_t)l * DD, xc, h_buf, l);
    }

    k_zero16<<<1, 32, 0, stream>>>(xlac);
    k_xlogits<<<(NNODES + 127) / 128, 128, 0, stream>>>(xc, Wxa, xlac);
    k_readout<<<dim3((NEDGES + 1 + 15) / 16, NBATCH), 256, 0, stream>>>(
        state, acb, xlac, bxa, Wr1, br1, Wr2, br2, out);
}
